// GNO_68238440399283
// MI455X (gfx1250) — compile-verified, matrix operand order checked by reading, not run on hardware
//
#include <hip/hip_runtime.h>
#include <hip/hip_bf16.h>

// ---------------------------------------------------------------------------
// Fused NNConv (edge-conditioned graph conv) for MI455X / gfx1250.
// Edge MLP (8->100->100->100->256) on v_wmma_f32_16x16x32_f16, fused with
// per-edge 16x16 apply + atomic scatter-mean. Intermediates never touch HBM.
// Each wave owns TWO M-tiles so every B-fragment load feeds two WMMAs.
// ---------------------------------------------------------------------------

typedef __attribute__((ext_vector_type(8)))  _Float16 v8h;
typedef __attribute__((ext_vector_type(16))) _Float16 v16h;
typedef __attribute__((ext_vector_type(8)))  float    v8f;

#define TILE_E   128     // edges per workgroup (8 M-tiles; 2 per wave)
#define NWAVES   4
#define BLOCK    (NWAVES * 32)
#define HPAD     128     // hidden 100 padded to 128 in K (x32 WMMA requirement)
#define NT_H     7       // N-tiles for hidden layers: cols 0..111 (100 valid)
#define KATTR    32      // attr K=8 padded to 32
#define WCOLS    256     // 16*16 per-edge weight matrix, flattened

// --- WMMA fragment loader (ISA 7.12.2, 16-bit A/B 16x32 layout) ------------
// lane 0..15  : row/col = lane,     elements 0..7 = K 0..7,   8..15 = K 16..23
// lane 16..31 : row/col = lane-16,  elements 0..7 = K 8..15,  8..15 = K 24..31
__device__ __forceinline__ v16h load_frag(const _Float16* base, int khalf) {
    v8h lo = *(const v8h*)(base + khalf * 8);
    v8h hi = *(const v8h*)(base + 16 + khalf * 8);
    v16h r;
#pragma unroll
    for (int i = 0; i < 8; ++i) { r[i] = lo[i]; r[8 + i] = hi[i]; }
    return r;
}

// One GEMM stage over TWO M-tiles per wave (B-fragment reuse):
//   [32 x (32*KS)] x [(32*KS) x (16*NT)]
// Activations from LDS (row-major f16), weights from global in transposed
// padded layout Wt[n][kpad] (f16), output -> LDS with fused bias (+ReLU), f16.
template <int NT, int KS, bool RELU>
__device__ __forceinline__ void gemm_stage(
    const _Float16* __restrict__ lds_in, int in_stride,
    const _Float16* __restrict__ Wt, int kpad,
    const float* __restrict__ bias, int nvalid,
    _Float16* __restrict__ lds_out, int out_stride,
    int ntile0, int lane, int mpair)
{
    const int rc    = lane & 15;   // row (A) or column (B/C)
    const int khalf = lane >> 4;
    const int m0    = mpair * 2;   // first of the two M-tiles

    v8f acc0[NT], acc1[NT];
#pragma unroll
    for (int t = 0; t < NT; ++t) {
        acc0[t] = (v8f){0.f,0.f,0.f,0.f,0.f,0.f,0.f,0.f};
        acc1[t] = (v8f){0.f,0.f,0.f,0.f,0.f,0.f,0.f,0.f};
    }

#pragma unroll
    for (int ks = 0; ks < KS; ++ks) {
        const int kb = ks * 32;
        v16h a0 = load_frag(lds_in + ((m0    ) * 16 + rc) * in_stride + kb, khalf);
        v16h a1 = load_frag(lds_in + ((m0 + 1) * 16 + rc) * in_stride + kb, khalf);
#pragma unroll
        for (int t = 0; t < NT; ++t) {
            v16h b = load_frag(Wt + ((ntile0 + t) * 16 + rc) * kpad + kb, khalf);
            acc0[t] = __builtin_amdgcn_wmma_f32_16x16x32_f16(
                false, a0, false, b, (short)0, acc0[t], false, false);
            acc1[t] = __builtin_amdgcn_wmma_f32_16x16x32_f16(
                false, a1, false, b, (short)0, acc1[t], false, false);
        }
    }

    // C layout: element r <-> row M = r + 8*khalf, column N = lane&15
    const int mrow0 = m0 * 16 + (khalf ? 8 : 0);
#pragma unroll
    for (int t = 0; t < NT; ++t) {
        const int  n     = (ntile0 + t) * 16 + rc;
        const bool valid = (n < nvalid);
        const float bv   = valid ? bias[n] : 0.0f;
#pragma unroll
        for (int r = 0; r < 8; ++r) {
            float v0 = acc0[t][r] + bv;
            float v1 = acc1[t][r] + bv;
            if (RELU) { v0 = v0 > 0.0f ? v0 : 0.0f; v1 = v1 > 0.0f ? v1 : 0.0f; }
            if (!valid) { v0 = 0.0f; v1 = 0.0f; }
            lds_out[(mrow0 + r     ) * out_stride + n] = (_Float16)v0;
            lds_out[(mrow0 + r + 16) * out_stride + n] = (_Float16)v1;
        }
    }
}

// --- weight prep: transpose + pad + f32->f16 -------------------------------
__global__ void prep_weights(const float* __restrict__ W1, const float* __restrict__ W2,
                             const float* __restrict__ W3, const float* __restrict__ W4,
                             _Float16* __restrict__ W1t, _Float16* __restrict__ W2t,
                             _Float16* __restrict__ W3t, _Float16* __restrict__ W4t)
{
    int tid = blockIdx.x * blockDim.x + threadIdx.x;
    const int s1 = 128 * KATTR;         // W1t: [128][32]
    const int s2 = s1 + 128 * HPAD;     // W2t: [128][128]
    const int s3 = s2 + 128 * HPAD;     // W3t: [128][128]
    const int s4 = s3 + 256 * HPAD;     // W4t: [256][128]
    if (tid < s1) {
        int n = tid / KATTR, k = tid % KATTR;
        W1t[tid] = (_Float16)((k < 8 && n < 100) ? W1[k * 100 + n] : 0.f);
    } else if (tid < s2) {
        int t = tid - s1; int n = t / HPAD, k = t % HPAD;
        W2t[t] = (_Float16)((k < 100 && n < 100) ? W2[k * 100 + n] : 0.f);
    } else if (tid < s3) {
        int t = tid - s2; int n = t / HPAD, k = t % HPAD;
        W3t[t] = (_Float16)((k < 100 && n < 100) ? W3[k * 100 + n] : 0.f);
    } else if (tid < s4) {
        int t = tid - s3; int n = t / HPAD, k = t % HPAD;
        W4t[t] = (_Float16)((k < 100 && n < 256) ? W4[k * 256 + n] : 0.f);
    }
}

__global__ void zero_f32(float* __restrict__ p, int n) {
    int i = blockIdx.x * blockDim.x + threadIdx.x;
    if (i < n) p[i] = 0.0f;
}

// --- fused edge kernel -----------------------------------------------------
__global__ __launch_bounds__(BLOCK)
void edge_kernel(const float* __restrict__ x,
                 const int*   __restrict__ edge_index,   // [2][E], int32
                 const float* __restrict__ edge_attr,    // [E][8]
                 const _Float16* __restrict__ W1t, const _Float16* __restrict__ W2t,
                 const _Float16* __restrict__ W3t, const _Float16* __restrict__ W4t,
                 const float* __restrict__ b1, const float* __restrict__ b2,
                 const float* __restrict__ b3, const float* __restrict__ b4,
                 float* __restrict__ agg, float* __restrict__ cnt, int E)
{
    __shared__ _Float16 s_attr[TILE_E * KATTR];   //  8 KB
    __shared__ _Float16 s_h[TILE_E * HPAD];       // 32 KB  (h1, then h3)
    __shared__ _Float16 s_w[TILE_E * WCOLS];      // 64 KB  (h2 in first half, then w)

    const int e0   = blockIdx.x * TILE_E;
    const int tid  = threadIdx.x;
    const int lane = tid & 31;
    const int wave = tid >> 5;

    // Zero h buffers once so pad columns (112..127) feed exact zeros into the
    // next stage's K dimension (WMMA propagates NaN even vs. zero weights).
    {
        const uint4 z4 = {0u, 0u, 0u, 0u};
        uint4* ph = (uint4*)s_h;
        uint4* pw = (uint4*)s_w;
        const int nvec = TILE_E * HPAD * (int)sizeof(_Float16) / 16;
        for (int i = tid; i < nvec; i += BLOCK) { ph[i] = z4; pw[i] = z4; }
    }

    // stage edge_attr tile into LDS as f16, K-padded to 32 with zeros
    {
        const int e  = tid;              // TILE_E == BLOCK
        const int ge = e0 + e;
        float4 a0 = {0.f,0.f,0.f,0.f}, a1 = {0.f,0.f,0.f,0.f};
        if (ge < E) {
            const float4* ea = (const float4*)edge_attr;
            a0 = ea[(long)ge * 2];
            a1 = ea[(long)ge * 2 + 1];
        }
        v8h h, z;
        h[0]=(_Float16)a0.x; h[1]=(_Float16)a0.y; h[2]=(_Float16)a0.z; h[3]=(_Float16)a0.w;
        h[4]=(_Float16)a1.x; h[5]=(_Float16)a1.y; h[6]=(_Float16)a1.z; h[7]=(_Float16)a1.w;
#pragma unroll
        for (int i = 0; i < 8; ++i) z[i] = (_Float16)0.f;
        v8h* dst = (v8h*)(s_attr + e * KATTR);
        dst[0] = h; dst[1] = z; dst[2] = z; dst[3] = z;
    }
    __syncthreads();

    // h1 = relu(attr @ W1 + b1)           [128 x 112]
    gemm_stage<NT_H, 1, true>(s_attr, KATTR, W1t, KATTR, b1, 100, s_h, HPAD, 0, lane, wave);
    __syncthreads();
    // h2 = relu(h1 @ W2 + b2)             [128 x 112]  -> first half of s_w
    gemm_stage<NT_H, 4, true>(s_h, HPAD, W2t, HPAD, b2, 100, s_w, HPAD, 0, lane, wave);
    __syncthreads();
    // h3 = relu(h2 @ W3 + b3)             [128 x 112]  -> back into s_h
    gemm_stage<NT_H, 4, true>(s_w, HPAD, W3t, HPAD, b3, 100, s_h, HPAD, 0, lane, wave);
    __syncthreads();
    // w  = h3 @ W4 + b4                   [128 x 256]  -> s_w (two N passes)
    gemm_stage<8, 4, false>(s_h, HPAD, W4t, HPAD, b4, 256, s_w, WCOLS, 0, lane, wave);
    gemm_stage<8, 4, false>(s_h, HPAD, W4t, HPAD, b4, 256, s_w, WCOLS, 8, lane, wave);
    __syncthreads();

    // per-edge apply + scatter: 1 thread per edge, 16 outputs
    const int ge = e0 + tid;
    if (ge < E) {
        const int s = edge_index[ge];
        const int d = edge_index[E + ge];
        const float* xs = x + (long)s * 16;
        const _Float16* we = s_w + tid * WCOLS;
        float acc[16];
#pragma unroll
        for (int j = 0; j < 16; ++j) acc[j] = 0.0f;
#pragma unroll
        for (int i = 0; i < 16; ++i) {
            const float xv = xs[i];
#pragma unroll
            for (int j = 0; j < 16; ++j) acc[j] += xv * (float)we[i * 16 + j];
        }
        float* ag = agg + (long)d * 16;
#pragma unroll
        for (int j = 0; j < 16; ++j) atomicAdd(&ag[j], acc[j]);
        atomicAdd(&cnt[d], 1.0f);
    }
}

// --- finalize: out = x @ root + agg/max(cnt,1) + bias ----------------------
__global__ void finalize_kernel(const float* __restrict__ x, const float* __restrict__ root,
                                const float* __restrict__ bias, const float* __restrict__ agg,
                                const float* __restrict__ cnt, float* __restrict__ out, int N)
{
    int n = blockIdx.x * blockDim.x + threadIdx.x;
    if (n >= N) return;
    float xi[16];
#pragma unroll
    for (int i = 0; i < 16; ++i) xi[i] = x[(long)n * 16 + i];
    float c = cnt[n];
    float inv = 1.0f / (c > 1.0f ? c : 1.0f);
#pragma unroll
    for (int o = 0; o < 16; ++o) {
        float acc = 0.0f;
#pragma unroll
        for (int i = 0; i < 16; ++i) acc += xi[i] * root[i * 16 + o];
        out[(long)n * 16 + o] = acc + agg[(long)n * 16 + o] * inv + bias[o];
    }
}

extern "C" void kernel_launch(void* const* d_in, const int* in_sizes, int n_in,
                              void* d_out, int out_size, void* d_ws, size_t ws_size,
                              hipStream_t stream)
{
    const float* x          = (const float*)d_in[0];
    const int*   edge_index = (const int*)  d_in[1];
    const float* edge_attr  = (const float*)d_in[2];
    const float* W1 = (const float*)d_in[3];  const float* b1 = (const float*)d_in[4];
    const float* W2 = (const float*)d_in[5];  const float* b2 = (const float*)d_in[6];
    const float* W3 = (const float*)d_in[7];  const float* b3 = (const float*)d_in[8];
    const float* W4 = (const float*)d_in[9];  const float* b4 = (const float*)d_in[10];
    const float* root = (const float*)d_in[11];
    const float* bias = (const float*)d_in[12];
    float* out = (float*)d_out;

    const int N = in_sizes[0] / 16;   // 50000
    const int E = in_sizes[1] / 2;    // 800000

    // workspace carve-up (f16 weight banks, then agg + cnt, contiguous)
    _Float16* W1t = (_Float16*)d_ws;                 // 128*32
    _Float16* W2t = W1t + 128 * KATTR;               // 128*128
    _Float16* W3t = W2t + 128 * HPAD;                // 128*128
    _Float16* W4t = W3t + 128 * HPAD;                // 256*128
    float*    agg = (float*)(W4t + 256 * HPAD);      // N*16
    float*    cnt = agg + (size_t)N * 16;            // N   (contiguous after agg)

    const int prep_total = 128 * KATTR + 128 * HPAD + 128 * HPAD + 256 * HPAD;
    prep_weights<<<(prep_total + 255) / 256, 256, 0, stream>>>(
        W1, W2, W3, W4, W1t, W2t, W3t, W4t);

    const int zn = N * 17;  // agg + cnt in one pass
    zero_f32<<<(zn + 255) / 256, 256, 0, stream>>>(agg, zn);

    edge_kernel<<<(E + TILE_E - 1) / TILE_E, BLOCK, 0, stream>>>(
        x, edge_index, edge_attr, W1t, W2t, W3t, W4t,
        b1, b2, b3, b4, agg, cnt, E);

    finalize_kernel<<<(N + 255) / 256, 256, 0, stream>>>(
        x, root, bias, agg, cnt, out, N);
}
